// RoIPooling_34660386078970
// MI455X (gfx1250) — matrix operand order repaired and checked by reading, not compiled, not run on hardware
//
#include <hip/hip_runtime.h>

#define B_DIM   2048
#define C_DIM   512
#define CROP    7
#define ROI     2
#define L_DIM   8
#define K_DIM   (L_DIM * C_DIM)   // 4096
#define OUT_DIM 4096

#define BM 128
#define BN 128
#define BK 32
#define LDS_STRIDE 40   // elements (80 bytes): 16B-aligned rows, conflict-free frag reads

typedef __attribute__((ext_vector_type(16))) __bf16 v16bf;
typedef __attribute__((ext_vector_type(8)))  float  v8f;

union Frag {
    uint4 q[2];
    v16bf v;
};

__device__ __forceinline__ unsigned short f32_to_bf16(float f) {
    union { float f; unsigned u; } v; v.f = f;
    unsigned u = v.u;
    unsigned r = u + 0x7FFFu + ((u >> 16) & 1u);   // round-to-nearest-even
    return (unsigned short)(r >> 16);
}

// ---------------- Kernel 1: W fp32 -> bf16 (vectorized x4) ----------------
__global__ void wconv_kernel(const float* __restrict__ W,
                             unsigned short* __restrict__ Wb) {
    int idx = blockIdx.x * blockDim.x + threadIdx.x;   // one float4 per thread
    float4 f = ((const float4*)W)[idx];
    ushort4 o;
    o.x = f32_to_bf16(f.x);
    o.y = f32_to_bf16(f.y);
    o.z = f32_to_bf16(f.z);
    o.w = f32_to_bf16(f.w);
    ((ushort4*)Wb)[idx] = o;
}

// ------------- Kernel 2: RoI 2x2 max-pool + bf16 convert ------------------
__global__ void pool_kernel(const float* __restrict__ feat,
                            const int* __restrict__ lms,
                            unsigned short* __restrict__ pooled) {
    int idx = blockIdx.x * blockDim.x + threadIdx.x;   // [0, B*L*C)
    int c = idx & (C_DIM - 1);
    int l = (idx >> 9) & (L_DIM - 1);
    int b = idx >> 12;

    int cx = lms[b * 2 * L_DIM + 2 * l + 0];
    int cy = lms[b * 2 * L_DIM + 2 * l + 1];
    int x1 = (int)((float)cx / 224.0f * 7.0f) - 1;
    int y1 = (int)((float)cy / 224.0f * 7.0f) - 1;
    x1 = x1 < 0 ? 0 : x1;
    y1 = y1 < 0 ? 0 : y1;
    x1 = x1 > (CROP - ROI) ? (CROP - ROI) : x1;
    y1 = y1 > (CROP - ROI) ? (CROP - ROI) : y1;
    if (cx == 0 && cy == 0) { x1 = 0; y1 = 0; }

    const float* p = feat + ((((size_t)b * C_DIM + c) * CROP + x1) * CROP + y1);
    float m0 = fmaxf(p[0],    p[1]);
    float m1 = fmaxf(p[CROP], p[CROP + 1]);
    pooled[idx] = f32_to_bf16(fmaxf(m0, m1));
}

// ------------- Kernel 3: bf16 WMMA GEMM  out = pooled @ W^T + b -----------
__global__ __launch_bounds__(256)
void gemm_kernel(const unsigned short* __restrict__ A,    // pooled [M,K] bf16 bits
                 const unsigned short* __restrict__ Bw,   // W      [N,K] bf16 bits
                 const float* __restrict__ bias,
                 float* __restrict__ Out) {
    __shared__ unsigned short As[BM * LDS_STRIDE];
    __shared__ unsigned short Bs[BN * LDS_STRIDE];

    const int tid  = threadIdx.x;
    const int lane = tid & 31;
    const int wave = tid >> 5;       // 8 waves
    const int wm   = wave >> 2;      // 0..1  -> 64 rows each
    const int wn   = wave & 3;       // 0..3  -> 32 cols each

    const int tile_m = blockIdx.y * BM;
    const int tile_n = blockIdx.x * BN;

    v8f acc[4][2] = {};

    // cooperative tile load: 512 x 16B chunks per operand tile, 2 per thread
    const int ch0 = tid;
    const int ch1 = tid + 256;
    const int r0 = ch0 >> 2, c0 = (ch0 & 3) * 8;
    const int r1 = ch1 >> 2, c1 = (ch1 & 3) * 8;

    const unsigned short* gA0 = A  + (size_t)(tile_m + r0) * K_DIM + c0;
    const unsigned short* gA1 = A  + (size_t)(tile_m + r1) * K_DIM + c1;
    const unsigned short* gB0 = Bw + (size_t)(tile_n + r0) * K_DIM + c0;
    const unsigned short* gB1 = Bw + (size_t)(tile_n + r1) * K_DIM + c1;

    for (int k0 = 0; k0 < K_DIM; k0 += BK) {
        uint4 a0 = *(const uint4*)(gA0 + k0);
        uint4 a1 = *(const uint4*)(gA1 + k0);
        uint4 b0 = *(const uint4*)(gB0 + k0);
        uint4 b1 = *(const uint4*)(gB1 + k0);
        if (k0 + BK < K_DIM) {           // hint next K-slab into cache
            __builtin_prefetch(gA0 + k0 + BK, 0, 0);
            __builtin_prefetch(gB0 + k0 + BK, 0, 0);
        }
        *(uint4*)(As + r0 * LDS_STRIDE + c0) = a0;
        *(uint4*)(As + r1 * LDS_STRIDE + c1) = a1;
        *(uint4*)(Bs + r0 * LDS_STRIDE + c0) = b0;
        *(uint4*)(Bs + r1 * LDS_STRIDE + c1) = b1;
        __syncthreads();

        // A fragments: ISA 16-bit A 16x32 layout
        //   lanes 0-15 : M=lane,    K 0-7 (q0) and 16-23 (q1)
        //   lanes 16-31: M=lane-16, K 8-15 (q0) and 24-31 (q1)
        const int rA    = lane & 15;
        const int halfA = (lane >> 4) * 8;    // element offset 0 or 8
        Frag afrag[4];
#pragma unroll
        for (int mi = 0; mi < 4; ++mi) {
            const unsigned short* src =
                As + (wm * 64 + mi * 16 + rA) * LDS_STRIDE + halfA;
            afrag[mi].q[0] = *(const uint4*)(src);
            afrag[mi].q[1] = *(const uint4*)(src + 16);
        }

        // B fragments: lane%16 = column N (a row of W)
        //   lanes 0-15: K 0-7 and 16-23, lanes 16-31: K 8-15 and 24-31
        const int nBr = lane & 15;
        const int kB  = (lane >> 4) * 8;
#pragma unroll
        for (int ni = 0; ni < 2; ++ni) {
            Frag bfrag;
            const unsigned short* src =
                Bs + (wn * 32 + ni * 16 + nBr) * LDS_STRIDE + kB;
            bfrag.q[0] = *(const uint4*)(src);
            bfrag.q[1] = *(const uint4*)(src + 16);
#pragma unroll
            for (int mi = 0; mi < 4; ++mi) {
                acc[mi][ni] = __builtin_amdgcn_wmma_f32_16x16x32_bf16(
                    false, afrag[mi].v, false, bfrag.v,
                    (short)0, acc[mi][ni], false, false);
            }
        }
        __syncthreads();
    }

    // Epilogue: C/D layout -> lane%16 = N, lanes>=16 carry M rows 8..15
    const int nOff  = lane & 15;
    const int mHalf = (lane >> 4) * 8;
#pragma unroll
    for (int ni = 0; ni < 2; ++ni) {
        int n = tile_n + wn * 32 + ni * 16 + nOff;
        float bv = bias[n];
#pragma unroll
        for (int mi = 0; mi < 4; ++mi) {
            int mbase = tile_m + wm * 64 + mi * 16 + mHalf;
#pragma unroll
            for (int r = 0; r < 8; ++r) {
                Out[(size_t)(mbase + r) * OUT_DIM + n] = acc[mi][ni][r] + bv;
            }
        }
    }
}

extern "C" void kernel_launch(void* const* d_in, const int* in_sizes, int n_in,
                              void* d_out, int out_size, void* d_ws, size_t ws_size,
                              hipStream_t stream) {
    const float* feat = (const float*)d_in[0];
    const int*   lms  = (const int*)d_in[1];
    const float* W    = (const float*)d_in[2];
    const float* bias = (const float*)d_in[3];
    float*       out  = (float*)d_out;

    unsigned short* Wb     = (unsigned short*)d_ws;                 // 4096*4096 bf16 (32 MB)
    unsigned short* pooled = Wb + (size_t)OUT_DIM * K_DIM;          // 2048*4096 bf16 (16 MB)

    // 1) convert W to bf16
    {
        int nvec = (OUT_DIM * K_DIM) / 4;
        wconv_kernel<<<nvec / 256, 256, 0, stream>>>(W, Wb);
    }
    // 2) RoI max-pool + bf16 convert
    {
        int total = B_DIM * L_DIM * C_DIM;
        pool_kernel<<<total / 256, 256, 0, stream>>>(feat, lms, pooled);
    }
    // 3) bf16 WMMA GEMM + bias
    {
        dim3 grid(OUT_DIM / BN, B_DIM / BM);
        gemm_kernel<<<grid, 256, 0, stream>>>(pooled, Wb, bias, out);
    }
}
